// Matching_43800076484878
// MI455X (gfx1250) — matrix operand order repaired and checked by reading, not compiled, not run on hardware
//
#include <hip/hip_runtime.h>
#include <hip/hip_bf16.h>
#include <math.h>

typedef __attribute__((ext_vector_type(16))) _Float16 v16h;
typedef __attribute__((ext_vector_type(8)))  float    v8f;

#define LEAK 0.1f
#define B_N 4
#define H_N 192
#define W_N 192
#define HW_N (H_N * W_N)
#define PIX_N (B_N * HW_N)

union ABu { v16h h; unsigned u[8]; uint4 q[2]; };
union Vec32h { _Float16 h[32]; uint4 q[4]; };

__device__ __forceinline__ int iminD(int a, int b) { return a < b ? a : b; }
__device__ __forceinline__ int imaxD(int a, int b) { return a > b ? a : b; }

// Async copy of 16 bytes global -> LDS (ASYNCcnt-tracked, no VGPR round trip).
// lds_off: byte offset inside this kernel's (single) static LDS allocation.
// goff:    byte offset from base (buffers are < 2^31 bytes).
__device__ __forceinline__ void async_g2l_b128(unsigned lds_off, const void* base, int goff)
{
    asm volatile("global_load_async_to_lds_b128 %0, %1, %2 offset:0"
                 :: "v"(lds_off), "v"(goff), "s"(base)
                 : "memory");
}
__device__ __forceinline__ void wait_asynccnt0()
{
    asm volatile("s_wait_asynccnt 0x0" ::: "memory");
}

// ---------------------------------------------------------------------------
// Implicit-GEMM convolution on v_wmma_f32_16x16x32_f16, LDS-staged activations.
//   Activations: NHWC f16, channels padded to Cinp (mult of 32, pads zeroed).
//   Weights:     f16 [CoutPad][KHW][Cinp] (A fragments contiguous).
//   Block = 8 waves = 8 rows x 16 cols of pixels. Per 32-channel chunk the
//   block stages the (8+2P)x(16+2P)x32 halo tile in LDS via
//   global_load_async_to_lds_b128 (zero ds_store at image borders), then each
//   wave reads B as 2 x ds_load_b128 per tap; A = 2 x global b128 per tile
//   per tap (identical across waves -> WGP$ broadcast).
// ---------------------------------------------------------------------------
template<int NT, int KW>
__global__ __launch_bounds__(256)
void conv_wmma_kernel(const _Float16* __restrict__ in, const _Float16* __restrict__ w,
                      const float* __restrict__ bias,
                      _Float16* __restrict__ out16, float* __restrict__ out32,
                      const float* __restrict__ add32,
                      int Cinp, int Cout, int ctot16, int choff16, int act)
{
    constexpr int P    = KW / 2;
    constexpr int TH2  = 8 + 2 * P;
    constexpr int TW2  = 16 + 2 * P;
    constexpr int KHW  = KW * KW;
    constexpr int NVEC = TH2 * TW2 * 4;     // uint4 vectors per 32-ch chunk

    __shared__ _Float16 tile[TH2 * TW2 * 32];

    const int tid  = threadIdx.x;
    const int lane = tid & 31;
    const int wv   = tid >> 5;
    const int n    = lane & 15;   // pixel column (B/D "N") and weight row (A "M")
    const int hi   = lane >> 4;
    const int kba  = hi * 16;     // A K base: lanes 0-15 -> K 0..15, 16-31 -> K 16..31
    const int kbb  = hi * 16;     // B K base: lanes 0-15 -> K 0..15, 16-31 -> K 16..31

    const int x0 = blockIdx.x * 16;
    const int y0 = blockIdx.y * 8;
    const int b  = blockIdx.z;
    const int y  = y0 + wv;
    const int x  = x0 + n;

    const int Kpad = KHW * Cinp;
    const _Float16* wbase = w + (size_t)n * Kpad;

    v8f zerov = {0.f, 0.f, 0.f, 0.f, 0.f, 0.f, 0.f, 0.f};
    v8f acc[NT];
#pragma unroll
    for (int t = 0; t < NT; ++t) acc[t] = zerov;

    for (int cc = 0; cc < Cinp; cc += 32) {
        __syncthreads();
        // stage halo tile for channels [cc, cc+32): async copies + zero borders
        for (int v = tid; v < NVEC; v += 256) {
            int ly = v / (TW2 * 4);
            int r  = v - ly * (TW2 * 4);
            int lx = r >> 2;
            int pt = r & 3;
            int gy = y0 + ly - P, gx = x0 + lx - P;
            unsigned lds_off = (unsigned)(((ly * TW2 + lx) * 32 + pt * 8) * 2);
            if (gy >= 0 && gy < H_N && gx >= 0 && gx < W_N) {
                int goff = (int)((((size_t)b * HW_N + (size_t)gy * W_N + gx) * Cinp
                                  + cc + pt * 8) * 2);
                async_g2l_b128(lds_off, in, goff);
            } else {
                *(uint4*)&tile[(ly * TW2 + lx) * 32 + pt * 8] = make_uint4(0u, 0u, 0u, 0u);
            }
        }
        wait_asynccnt0();
        __syncthreads();

#pragma unroll
        for (int khw = 0; khw < KHW; ++khw) {
            const int kh  = khw / KW;
            const int kwv = khw % KW;
            const _Float16* bt = &tile[((wv + kh) * TW2 + (n + kwv)) * 32 + kbb];
            ABu Bm;
            Bm.q[0] = *(const uint4*)(bt);         // K (cc+kbb)+0..7
            Bm.q[1] = *(const uint4*)(bt + 8);     // K (cc+kbb)+8..15
            const _Float16* arow = wbase + (size_t)khw * Cinp + cc;
#pragma unroll
            for (int t = 0; t < NT; ++t) {
                ABu A;
                const _Float16* ap = arow + (size_t)t * 16 * Kpad + kba;
                A.q[0] = *(const uint4*)(ap);      // K (cc+kba)+0..7
                A.q[1] = *(const uint4*)(ap + 8);  // K (cc+kba)+8..15
                acc[t] = __builtin_amdgcn_wmma_f32_16x16x32_f16(
                    false, A.h, false, Bm.h, (short)0, acc[t], false, false);
            }
        }
    }

    const size_t opix = (size_t)b * HW_N + (size_t)y * W_N + x;
#pragma unroll
    for (int t = 0; t < NT; ++t) {
#pragma unroll
        for (int v = 0; v < 8; ++v) {
            int c = t * 16 + v + 8 * hi;    // D layout: M = v + 8*(lane>=16)
            if (c >= Cout) continue;
            float val = acc[t][v] + (bias ? bias[c] : 0.f);
            if (act == 1)      val = (val >= 0.f) ? val : LEAK * val;
            else if (act == 2) val = 1.f / (1.f + __expf(-val));
            if (out16)
                out16[opix * ctot16 + choff16 + c] = (_Float16)val;
            if (out32) {
                size_t o = ((size_t)b * Cout + c) * HW_N + (size_t)y * W_N + x;
                float a2 = add32 ? add32[o] : 0.f;
                out32[o] = val + a2;
            }
        }
    }
}

// ---------------------------------------------------------------------------
// LDS-tiled correlation on NHWC f16 inputs.
//   out[b,d,y,x] = lrelu( (1/96) * sum_c f1[pix][c] * f2[pix+d][c] )
//   Block = 8x16 pixels, 256 threads (2 threads/pixel split the disparities).
//   Per 32-channel chunk: stage (8+2MD)x(16+2MD)x32 f2 halo in LDS, keep own
//   f1 chunk in registers, accumulate all disparities.
// ---------------------------------------------------------------------------
template<int MD, int STRIDE>
__global__ __launch_bounds__(256)
void corr_tiled_kernel(const _Float16* __restrict__ f1, int s1,
                       const _Float16* __restrict__ f2, int s2,
                       _Float16* __restrict__ outA, int ctotA, int choffA,
                       _Float16* __restrict__ outB, int ctotB)
{
    constexpr int ND   = 2 * MD / STRIDE + 1;
    constexpr int ND2  = ND * ND;
    constexpr int DH   = (ND2 + 1) / 2;       // disps per thread
    constexpr int TH2  = 8 + 2 * MD;
    constexpr int TW2  = 16 + 2 * MD;
    constexpr int NVEC = TH2 * TW2 * 4;       // uint4 per 32-ch chunk

    __shared__ _Float16 tile[TH2 * TW2 * 32];

    const int tid = threadIdx.x;
    const int x0  = blockIdx.x * 16;
    const int y0  = blockIdx.y * 8;
    const int b   = blockIdx.z;

    const int p  = tid & 127;      // pixel within 8x16 tile
    const int hf = tid >> 7;       // disparity half
    const int ly = p >> 4;
    const int lx = p & 15;
    const int y  = y0 + ly;
    const int x  = x0 + lx;
    const int d0 = hf * DH;

    float accd[DH];
#pragma unroll
    for (int di = 0; di < DH; ++di) accd[di] = 0.f;

    const _Float16* f1p = f1 + ((size_t)b * HW_N + (size_t)y * W_N + x) * s1;

    for (int cc = 0; cc < 96; cc += 32) {
        __syncthreads();
        for (int v = tid; v < NVEC; v += 256) {
            int site = v >> 2;
            int pt   = v & 3;
            int lyt  = site / TW2;
            int lxt  = site - lyt * TW2;
            int gy = y0 + lyt - MD, gx = x0 + lxt - MD;
            uint4 val = make_uint4(0u, 0u, 0u, 0u);
            if (gy >= 0 && gy < H_N && gx >= 0 && gx < W_N)
                val = *(const uint4*)(f2 + ((size_t)b * HW_N + (size_t)gy * W_N + gx) * s2
                                      + cc + pt * 8);
            *(uint4*)&tile[site * 32 + pt * 8] = val;
        }
        __syncthreads();

        Vec32h av;
#pragma unroll
        for (int pt = 0; pt < 4; ++pt)
            av.q[pt] = *(const uint4*)(f1p + cc + pt * 8);

#pragma unroll
        for (int di = 0; di < DH; ++di) {
            int d = d0 + di;
            if (d >= ND2) break;
            int dy = (d / ND) * STRIDE - MD;
            int dx = (d % ND) * STRIDE - MD;
            const _Float16* bp = &tile[(((ly + MD + dy) * TW2) + (lx + MD + dx)) * 32];
            Vec32h bv;
#pragma unroll
            for (int pt = 0; pt < 4; ++pt)
                bv.q[pt] = *(const uint4*)(bp + pt * 8);
            float acc = 0.f;
#pragma unroll
            for (int c = 0; c < 32; ++c)
                acc += (float)av.h[c] * (float)bv.h[c];
            accd[di] += acc;
        }
    }

    const size_t pixo = (size_t)b * HW_N + (size_t)y * W_N + x;
#pragma unroll
    for (int di = 0; di < DH; ++di) {
        int d = d0 + di;
        if (d >= ND2) break;
        float vv = accd[di] * (1.f / 96.f);
        vv = (vv >= 0.f) ? vv : LEAK * vv;
        _Float16 h = (_Float16)vv;
        outA[pixo * ctotA + choffA + d] = h;
        if (outB) outB[pixo * ctotB + d] = h;
    }
}

// ---------------------------------------------------------------------------
// Weight pack: f32 OIHW -> f16 [CoutPad][KHW][Cinp] (zero padded).
// ---------------------------------------------------------------------------
__global__ void pack_w_kernel(const float* __restrict__ src, _Float16* __restrict__ dst,
                              int Cout, int Cin, int KHW, int Cinp, long total)
{
    long i = (long)blockIdx.x * blockDim.x + threadIdx.x;
    if (i >= total) return;
    int Kpad = KHW * Cinp;
    int r    = (int)(i / Kpad);
    int rem  = (int)(i - (long)r * Kpad);
    int khw  = rem / Cinp;
    int cin  = rem - khw * Cinp;
    float v = (r < Cout && cin < Cin) ? src[((long)r * Cin + cin) * KHW + khw] : 0.f;
    dst[i] = (_Float16)v;
}

__global__ void zero16_kernel(_Float16* __restrict__ p, long n)
{
    long i = (long)blockIdx.x * blockDim.x + threadIdx.x;
    if (i < n) p[i] = (_Float16)0.f;
}

// f32 NCHW -> f16 NHWC slice of a concat buffer.
__global__ void nchw_to_nhwc_f16_kernel(const float* __restrict__ src, _Float16* __restrict__ dst,
                                        int Csrc, int ctot, int choff, long total)
{
    long i = (long)blockIdx.x * blockDim.x + threadIdx.x;
    if (i >= total) return;
    int  c   = (int)(i % Csrc);
    long pix = i / Csrc;
    int  b   = (int)(pix / HW_N);
    int  pos = (int)(pix - (long)b * HW_N);
    dst[(size_t)pix * ctot + choff + c] = (_Float16)src[((long)b * Csrc + c) * HW_N + pos];
}

// ---------------------------------------------------------------------------
// 2x transposed conv (grouped, 4x4, stride 2, pad 2, flipped weights). 96->192.
// ---------------------------------------------------------------------------
__global__ void deconv2x_kernel(const float* __restrict__ xin, const float* __restrict__ w4,
                                float* __restrict__ o32, _Float16* __restrict__ o16,
                                int ctot, int choff, int G, int Hi, int Wi, long total)
{
    long i = (long)blockIdx.x * blockDim.x + threadIdx.x;
    if (i >= total) return;
    int  x = (int)(i % W_N);
    long t = i / W_N;
    int  y = (int)(t % H_N); t /= H_N;
    int  g = (int)(t % G);
    int  b = (int)(t / G);

    float acc = 0.f;
#pragma unroll
    for (int kh = 0; kh < 4; ++kh) {
        int iy = y + kh - 2;
        if (iy < 0 || iy > 2 * Hi - 2 || (iy & 1)) continue;
#pragma unroll
        for (int kw = 0; kw < 4; ++kw) {
            int ix = x + kw - 2;
            if (ix < 0 || ix > 2 * Wi - 2 || (ix & 1)) continue;
            acc += w4[g * 16 + (3 - kh) * 4 + (3 - kw)] *
                   xin[(((long)b * G + g) * Hi + (iy >> 1)) * Wi + (ix >> 1)];
        }
    }
    if (o32) o32[(((long)b * G + g) * H_N + y) * W_N + x] = acc;
    if (o16) o16[((size_t)b * HW_N + (size_t)y * W_N + x) * ctot + choff + g] = (_Float16)acc;
}

// ---------------------------------------------------------------------------
// Bilinear backwarp. Input NCHW f32, flow NCHW f32 (2ch).
// Output: NCHW f32 (out32) or NHWC f16 with ostride (out16).
// ---------------------------------------------------------------------------
__global__ void backwarp_kernel(const float* __restrict__ inp, const float* __restrict__ flw,
                                float scale, float* __restrict__ out32,
                                _Float16* __restrict__ out16, int ostride, int C)
{
    int i = blockIdx.x * blockDim.x + threadIdx.x;
    if (i >= PIX_N) return;
    int x = i % W_N;
    int t = i / W_N;
    int y = t % H_N;
    int b = t / H_N;

    float fx = flw[(((long)b * 2 + 0) * H_N + y) * W_N + x] * scale;
    float fy = flw[(((long)b * 2 + 1) * H_N + y) * W_N + x] * scale;
    float gx = (2.f * x + 1.f) / (float)W_N - 1.f + fx / ((W_N - 1.f) * 0.5f);
    float gy = (2.f * y + 1.f) / (float)H_N - 1.f + fy / ((H_N - 1.f) * 0.5f);
    float xs = ((gx + 1.f) * W_N - 1.f) * 0.5f;
    float ys = ((gy + 1.f) * H_N - 1.f) * 0.5f;

    float x0 = floorf(xs), y0 = floorf(ys);
    float w00 = (x0 + 1.f - xs) * (y0 + 1.f - ys);
    float w10 = (xs - x0) * (y0 + 1.f - ys);
    float w01 = (x0 + 1.f - xs) * (ys - y0);
    float w11 = (xs - x0) * (ys - y0);

    int xi = (int)x0, yi = (int)y0;
    bool vx0 = (xi     >= 0 && xi     < W_N), vx1 = (xi + 1 >= 0 && xi + 1 < W_N);
    bool vy0 = (yi     >= 0 && yi     < H_N), vy1 = (yi + 1 >= 0 && yi + 1 < H_N);
    int xc0 = iminD(imaxD(xi,     0), W_N - 1), xc1 = iminD(imaxD(xi + 1, 0), W_N - 1);
    int yc0 = iminD(imaxD(yi,     0), H_N - 1), yc1 = iminD(imaxD(yi + 1, 0), H_N - 1);

    for (int c = 0; c < C; ++c) {
        const float* base = inp + ((long)b * C + c) * (long)HW_N;
        float v00 = (vx0 && vy0) ? base[yc0 * W_N + xc0] : 0.f;
        float v10 = (vx1 && vy0) ? base[yc0 * W_N + xc1] : 0.f;
        float v01 = (vx0 && vy1) ? base[yc1 * W_N + xc0] : 0.f;
        float v11 = (vx1 && vy1) ? base[yc1 * W_N + xc1] : 0.f;
        float r = v00 * w00 + v10 * w10 + v01 * w01 + v11 * w11;
        if (out32) out32[((long)b * C + c) * (long)HW_N + y * W_N + x] = r;
        if (out16) out16[((size_t)b * HW_N + (size_t)y * W_N + x) * ostride + c] = (_Float16)r;
    }
}

// corr = cs * corr + co  (NHWC f16, ctot-strided, real channels only, in place)
__global__ void modulate_kernel(_Float16* __restrict__ corr, const _Float16* __restrict__ cs,
                                const _Float16* __restrict__ co, int Creal, int ctot, long total)
{
    long i = (long)blockIdx.x * blockDim.x + threadIdx.x;
    if (i >= total) return;
    int  c   = (int)(i % Creal);
    long pix = i / Creal;
    size_t idx = (size_t)pix * ctot + c;
    float cv = (float)corr[idx], s = (float)cs[idx], o = (float)co[idx];
    corr[idx] = (_Float16)(s * cv + o);
}

// ---------------------------------------------------------------------------
// Host side
// ---------------------------------------------------------------------------
struct LayerDef { int cin, cout, kh, pad; };

static void run_conv(hipStream_t st, const _Float16* in, const _Float16* w, const float* bias,
                     _Float16* o16, int ctot, int choff, float* o32, const float* add32,
                     int Cinp, int Cout, int KH, int act)
{
    const int nt = (Cout + 15) / 16;
    dim3 grid(W_N / 16, H_N / 8, B_N);
    dim3 blk(256);
#define CC(NTV, KWV) conv_wmma_kernel<NTV, KWV><<<grid, blk, 0, st>>>( \
        in, w, bias, o16, o32, add32, Cinp, Cout, ctot, choff, act)
    if (KH == 3) {
        switch (nt) {
            case 2: CC(2, 3); break;
            case 4: CC(4, 3); break;
            case 6: CC(6, 3); break;
            case 8: CC(8, 3); break;
            default: break;
        }
    } else if (KH == 5) {
        if (nt == 1) CC(1, 5);
    } else if (KH == 1) {
        if (nt == 6) CC(6, 1);
    }
#undef CC
}

static inline int rup32(int v) { return (v + 31) / 32 * 32; }

extern "C" void kernel_launch(void* const* d_in, const int* in_sizes, int n_in,
                              void* d_out, int out_size, void* d_ws, size_t ws_size,
                              hipStream_t stream)
{
    (void)in_sizes; (void)n_in; (void)out_size; (void)ws_size;
    hipStream_t st = stream;

    const float* feat1    = (const float*)d_in[2];
    const float* feat2    = (const float*)d_in[3];
    const float* flowIn   = (const float*)d_in[4];
    const float* confIn   = (const float*)d_in[5];
    const float* upconf_w = (const float*)d_in[6];
    const float* upflow_w = (const float*)d_in[7];

    // params-dict order: cf1,cf2,cf3,disp,confnet,crf1,crf2,cs1,cs2,co1,co2,m1..m6
    const LayerDef ls[17] = {
        {50, 128, 3, 1},  // 0 cf1
        {128, 64, 3, 1},  // 1 cf2
        {64,  32, 3, 1},  // 2 cf3
        {32,   2, 5, 2},  // 3 disp
        {32,   1, 5, 2},  // 4 confnet
        {178,128, 3, 1},  // 5 crf1
        {128, 64, 3, 1},  // 6 crf2
        {64,  32, 3, 1},  // 7 cs1
        {32,  81, 1, 0},  // 8 cs2
        {64,  32, 3, 1},  // 9 co1
        {32,  81, 1, 0},  // 10 co2
        {81, 128, 3, 1},  // 11 m1
        {128,128, 3, 1},  // 12 m2
        {128, 96, 3, 1},  // 13 m3
        {96,  64, 3, 1},  // 14 m4
        {64,  32, 3, 1},  // 15 m5
        {32,   2, 5, 2},  // 16 m6
    };
    const float* Wc[17]; const float* Bc[17];
    for (int i = 0; i < 17; ++i) {
        Wc[i] = (const float*)d_in[8 + 2 * i];
        Bc[i] = (const float*)d_in[8 + 2 * i + 1];
    }

    // bump allocator over workspace
    char*  ws  = (char*)d_ws;
    size_t off = 0;
    auto alloc = [&](size_t bytes) -> void* {
        void* p = ws + off;
        off = (off + bytes + 255) & ~(size_t)255;
        return p;
    };
    const size_t c16 = (size_t)PIX_N * 2;
    const size_t c32 = (size_t)PIX_N * 4;

    // NHWC f16 activation buffers (channel counts padded to mult of 32)
    _Float16* cat64  = (_Float16*)alloc(64  * c16);  // [49 self-corr | 1 conf | 14 zero]
    _Float16* cat192 = (_Float16*)alloc(192 * c16);  // [96 feat1 | 81 corr | 1 conf | 14 zero]
    _Float16* x128a  = (_Float16*)alloc(128 * c16);
    _Float16* x128b  = (_Float16*)alloc(128 * c16);
    _Float16* x96    = (_Float16*)alloc(96  * c16);
    _Float16* x64    = (_Float16*)alloc(64  * c16);
    _Float16* x32a   = (_Float16*)alloc(32  * c16);
    _Float16* x32b   = (_Float16*)alloc(32  * c16);
    _Float16* corr96 = (_Float16*)alloc(96  * c16);
    _Float16* cs96   = (_Float16*)alloc(96  * c16);
    _Float16* co96   = (_Float16*)alloc(96  * c16);
    _Float16* f2w16  = (_Float16*)alloc(96  * c16);  // warped feat2, NHWC f16
    float* dispB  = (float*)alloc(2 * c32);
    float* flowUp = (float*)alloc(2 * c32);
    float* flowW  = (float*)alloc(2 * c32);

    // zero the buffers whose pad channels are consumed by WMMA convs
    {
        long n;
        n = 64  * (long)PIX_N; zero16_kernel<<<(unsigned)((n+255)/256),256,0,st>>>(cat64,  n);
        n = 192 * (long)PIX_N; zero16_kernel<<<(unsigned)((n+255)/256),256,0,st>>>(cat192, n);
        n = 96  * (long)PIX_N; zero16_kernel<<<(unsigned)((n+255)/256),256,0,st>>>(corr96, n);
    }

    // pack all conv weights to f16 [CoutPad][KHW][Cinp]
    _Float16* Wh[17]; int Cip[17];
    for (int i = 0; i < 17; ++i) {
        int KHW = ls[i].kh * ls[i].kh;
        Cip[i]  = rup32(ls[i].cin);
        int CoP = (ls[i].cout + 15) / 16 * 16;
        long tot = (long)CoP * KHW * Cip[i];
        Wh[i] = (_Float16*)alloc((size_t)tot * 2);
        pack_w_kernel<<<(unsigned)((tot + 255) / 256), 256, 0, st>>>(
            Wc[i], Wh[i], ls[i].cout, ls[i].cin, KHW, Cip[i], tot);
    }

    float* outFlow = (float*)d_out;
    float* outConf = outFlow + (size_t)B_N * 2 * HW_N;

    dim3 tgrid(W_N / 16, H_N / 8, B_N);

    // --- cf branch ------------------------------------------------------
    { // feat1 -> cat192 ch 0..95
        long tot = (long)PIX_N * 96;
        nchw_to_nhwc_f16_kernel<<<(unsigned)((tot + 255) / 256), 256, 0, st>>>(
            feat1, cat192, 96, 192, 0, tot);
    }
    { // conf = deconv2x(conf) -> cat64 ch 49
        long tot = (long)B_N * 1 * HW_N;
        deconv2x_kernel<<<(unsigned)((tot + 255) / 256), 256, 0, st>>>(
            confIn, upconf_w, nullptr, cat64, 64, 49, 1, 96, 96, tot);
    }
    // ac = lrelu(correlation(feat1, feat1, 6, 2)) -> cat64 ch 0..48
    corr_tiled_kernel<6, 2><<<tgrid, 256, 0, st>>>(cat192, 192, cat192, 192,
                                                   cat64, 64, 0, nullptr, 0);
    run_conv(st, cat64, Wh[0], Bc[0], x128a, 128, 0, nullptr, nullptr, Cip[0], 128, 3, 1); // cf1
    run_conv(st, x128a, Wh[1], Bc[1], x64,   64,  0, nullptr, nullptr, Cip[1], 64,  3, 1); // cf2
    run_conv(st, x64,   Wh[2], Bc[2], x32a,  32,  0, nullptr, nullptr, Cip[2], 32,  3, 1); // cf3
    run_conv(st, x32a,  Wh[3], Bc[3], nullptr, 0, 0, dispB,  nullptr,  Cip[3], 2,   5, 0); // disp
    run_conv(st, x32a,  Wh[4], Bc[4], cat192, 192, 177, outConf, nullptr, Cip[4], 1, 5, 2); // confnet

    // --- flow warp ------------------------------------------------------
    { // flow = deconv2x(flow)
        long tot = (long)B_N * 2 * HW_N;
        deconv2x_kernel<<<(unsigned)((tot + 255) / 256), 256, 0, st>>>(
            flowIn, upflow_w, flowUp, nullptr, 0, 0, 2, 96, 96, tot);
    }
    // flow = backwarp(flowUp, disp) -> flowW (f32 NCHW)
    backwarp_kernel<<<(PIX_N + 255) / 256, 256, 0, st>>>(flowUp, dispB, 1.0f,
                                                         flowW, nullptr, 0, 2);
    // feat2w = backwarp(feat2, flow*2.5) -> NHWC f16
    backwarp_kernel<<<(PIX_N + 255) / 256, 256, 0, st>>>(feat2, flowW, 2.5f,
                                                         nullptr, f2w16, 96, 96);
    // corr = lrelu(correlation(feat1, feat2w, 4, 1)) -> cat192 ch 96..176 + corr96
    corr_tiled_kernel<4, 1><<<tgrid, 256, 0, st>>>(cat192, 192, f2w16, 96,
                                                   cat192, 192, 96, corr96, 96);

    // --- crf / modulation ----------------------------------------------
    run_conv(st, cat192, Wh[5],  Bc[5],  x128a, 128, 0, nullptr, nullptr, Cip[5],  128, 3, 1); // crf1
    run_conv(st, x128a,  Wh[6],  Bc[6],  x64,   64,  0, nullptr, nullptr, Cip[6],  64,  3, 1); // crf2
    run_conv(st, x64,    Wh[7],  Bc[7],  x32a,  32,  0, nullptr, nullptr, Cip[7],  32,  3, 1); // cs1
    run_conv(st, x32a,   Wh[8],  Bc[8],  cs96,  96,  0, nullptr, nullptr, Cip[8],  81,  1, 0); // cs2
    run_conv(st, x64,    Wh[9],  Bc[9],  x32b,  32,  0, nullptr, nullptr, Cip[9],  32,  3, 1); // co1
    run_conv(st, x32b,   Wh[10], Bc[10], co96,  96,  0, nullptr, nullptr, Cip[10], 81,  1, 0); // co2
    { // corr = cs * corr + co
        long tot = (long)PIX_N * 81;
        modulate_kernel<<<(unsigned)((tot + 255) / 256), 256, 0, st>>>(corr96, cs96, co96, 81, 96, tot);
    }

    // --- m tower --------------------------------------------------------
    run_conv(st, corr96, Wh[11], Bc[11], x128a, 128, 0, nullptr, nullptr, Cip[11], 128, 3, 1); // m1
    run_conv(st, x128a,  Wh[12], Bc[12], x128b, 128, 0, nullptr, nullptr, Cip[12], 128, 3, 1); // m2
    run_conv(st, x128b,  Wh[13], Bc[13], x96,   96,  0, nullptr, nullptr, Cip[13], 96,  3, 1); // m3
    run_conv(st, x96,    Wh[14], Bc[14], x64,   64,  0, nullptr, nullptr, Cip[14], 64,  3, 1); // m4
    run_conv(st, x64,    Wh[15], Bc[15], x32a,  32,  0, nullptr, nullptr, Cip[15], 32,  3, 1); // m5
    // m6 (no act), fused residual: out_flow = flowW + m6
    run_conv(st, x32a,   Wh[16], Bc[16], nullptr, 0, 0, outFlow, flowW,   Cip[16], 2,   5, 0); // m6
}
